// GlobalPatchAttention_23914377904860
// MI455X (gfx1250) — matrix operand order repaired and checked by reading, compiled, not run on hardware
//
#include <hip/hip_runtime.h>
#include <hip/hip_bf16.h>

#define NP    2048
#define DT    256
#define PD    2048
#define BATCH 8
#define MTOT  (BATCH * NP)   // 16384

typedef __attribute__((ext_vector_type(16))) __bf16 v16bf;
typedef __attribute__((ext_vector_type(8)))  __bf16 v8bf;
typedef __attribute__((ext_vector_type(4)))  __bf16 v4bf;
typedef __attribute__((ext_vector_type(8)))  float  v8f;
typedef __attribute__((ext_vector_type(4)))  unsigned int v4u;
typedef __attribute__((ext_vector_type(8)))  int v8i;
typedef __attribute__((ext_vector_type(4)))  int v4i;

#if __has_builtin(__builtin_amdgcn_tensor_load_to_lds) && __has_builtin(__builtin_amdgcn_s_wait_tensorcnt)
#define GPA_TDM 1
#else
#define GPA_TDM 0
#endif

__device__ __forceinline__ v8f wmma_bf16(v16bf a, v16bf b, v8f c) {
  return __builtin_amdgcn_wmma_f32_16x16x32_bf16(false, a, false, b, (short)0, c, false, false);
}

// A-tile (16 rows x 32 K) from row-major bf16, leading dim lda (elements).
__device__ __forceinline__ v16bf load_tile_a(const __bf16* __restrict__ p, int lda, int lane) {
  const int row = lane & 15;
  const int ko  = (lane >> 4) << 3;
  const __bf16* q = p + row * lda + ko;
  v8bf lo = *reinterpret_cast<const v8bf*>(q);
  v8bf hi = *reinterpret_cast<const v8bf*>(q + 16);
  return __builtin_shufflevector(lo, hi, 0,1,2,3,4,5,6,7,8,9,10,11,12,13,14,15);
}

// B-tile (32 K x 16 N) from N-major bf16 (row n holds contiguous K), leading dim ldb.
__device__ __forceinline__ v16bf load_tile_b(const __bf16* __restrict__ p, int ldb, int lane) {
  const int n  = lane & 15;
  const int ko = (lane >> 4) << 3;
  const __bf16* q = p + n * ldb + ko;
  v8bf lo = *reinterpret_cast<const v8bf*>(q);
  v8bf hi = *reinterpret_cast<const v8bf*>(q + 16);
  return __builtin_shufflevector(lo, hi, 0,1,2,3,4,5,6,7,8,9,10,11,12,13,14,15);
}

__device__ __forceinline__ float gate(float x) {           // exp(sigmoid(x)/16)
  float sg = __frcp_rn(1.0f + __expf(-x));
  return __expf(sg * 0.0625f);
}

__device__ __forceinline__ unsigned lds_off(const void* p) {  // flat addr low 32 bits = LDS offset
  return (unsigned)(unsigned long long)(uintptr_t)p;
}

#if GPA_TDM
// 2-D TDM load: tile (tile_d0 x tile_d1) bf16 elements, global row stride0 (elements),
// from global tile start -> contiguous LDS rows of tile_d0 elements. ISA 8.3/8.4 D# layout.
__device__ __forceinline__ void tdm_load_2d(unsigned ldsoff, const void* gaddr,
                                            unsigned tile_d0, unsigned tile_d1,
                                            unsigned tensor_d0, unsigned tensor_d1,
                                            unsigned long long stride0) {
  unsigned long long ga = (unsigned long long)(uintptr_t)gaddr;
  v4u g0;
  g0[0] = 1u;                                             // count=1 (valid user D#)
  g0[1] = ldsoff;                                         // lds_addr (bytes)
  g0[2] = (unsigned)(ga & 0xffffffffu);                   // global_addr[31:0]
  g0[3] = (unsigned)((ga >> 32) & 0x01ffffffu) | (2u << 30); // global_addr[56:32] | type=2
  v8i g1;
  g1[0] = (int)(1u << 16);                                // data_size = 2 bytes
  g1[1] = (int)((tensor_d0 & 0xffffu) << 16);             // tensor_dim0[15:0]
  g1[2] = (int)((tensor_d0 >> 16) | ((tensor_d1 & 0xffffu) << 16));
  g1[3] = (int)((tensor_d1 >> 16) | (tile_d0 << 16));     // tile_dim0
  g1[4] = (int)(tile_d1 & 0xffffu);                       // tile_dim1 (tile_dim2 = 0)
  g1[5] = (int)(unsigned)(stride0 & 0xffffffffu);         // tensor_dim0_stride[31:0]
  g1[6] = (int)(unsigned)((stride0 >> 32) & 0xffffu);     // [47:32] (tensor_dim1_stride = 0)
  g1[7] = 0;
  v4i z4 = {0, 0, 0, 0};
  v8i z8 = {0, 0, 0, 0, 0, 0, 0, 0};
  __builtin_amdgcn_tensor_load_to_lds(g0, g1, z4, z4, z8, 0);  // 6-arg toolchain form
}
#define TDM_WAIT(n) __builtin_amdgcn_s_wait_tensorcnt(n)       // n must be a literal
#endif

// Cooperative fallback copy (all 256 threads): strided rows -> packed LDS rows.
__device__ __forceinline__ void coop_copy_rows(__bf16* __restrict__ dst, const __bf16* __restrict__ src,
                                               int rows, int rowElems, int srcLd, int tid, int nthr) {
  const int chunksPerRow = rowElems >> 3;                  // 16B chunks
  for (int idx = tid; idx < rows * chunksPerRow; idx += nthr) {
    int r = idx / chunksPerRow, c = idx - r * chunksPerRow;
    *reinterpret_cast<v8bf*>(dst + r * rowElems + c * 8) =
        *reinterpret_cast<const v8bf*>(src + (size_t)r * srcLd + c * 8);
  }
}

// ---------------------------------------------------------------- convert x -> bf16
__global__ __launch_bounds__(256) void gpa_cvt_kernel(const float4* __restrict__ x,
                                                      v4bf* __restrict__ y, int n4) {
  int i = blockIdx.x * 256 + threadIdx.x;
  if (i >= n4) return;
  float4 f = x[i];
  v4bf o;
  o[0] = (__bf16)f.x; o[1] = (__bf16)f.y; o[2] = (__bf16)f.z; o[3] = (__bf16)f.w;
  y[i] = o;
}

// ---------------------------------------------------------------- W [PD,DT] -> Wt [DT,PD] bf16
__global__ __launch_bounds__(256) void gpa_wtrans_kernel(const float* __restrict__ W,
                                                         __bf16* __restrict__ Wt) {
  int idx = blockIdx.x * 256 + threadIdx.x;   // n*PD + k
  int n = idx >> 11;
  int k = idx & (PD - 1);
  Wt[idx] = (__bf16)W[(size_t)k * DT + n];
}

// ---------------------------------------------------------------- QKV projection
// Block: 8 waves, same weight matrix o; wave -> 16-row tile. Weight slab (256 x 32 K)
// double-buffered in LDS via TDM; A-tile global load pipelined one K-step ahead.
__global__ __launch_bounds__(256) void gpa_qkv_kernel(
    const __bf16* __restrict__ xb,   // [MTOT, PD]
    const __bf16* __restrict__ wt,   // [3][DT, PD] N-major
    const float* __restrict__ bqv, const float* __restrict__ bkv, const float* __restrict__ bvv,
    __bf16* __restrict__ qb, __bf16* __restrict__ kb, __bf16* __restrict__ vT)
{
  __shared__ __bf16 wbuf[2][DT * 32];          // 2 x 16 KB
  const int tid  = threadIdx.x;
  const int lane = tid & 31;
  const int wv   = tid >> 5;
  const int wv_s = __builtin_amdgcn_readfirstlane(tid) >> 5;   // scalar wave id
  const int wid  = blockIdx.x * 8 + wv;        // 0..3071
  const int o    = wid >> 10;                  // uniform per block
  const int m0   = (wid & 1023) << 4;

  const __bf16* A  = xb + (size_t)m0 * PD;
  const __bf16* Bt = wt + (size_t)o * DT * PD;

#if GPA_TDM
  if (wv_s == 0)
    tdm_load_2d(lds_off(&wbuf[0][0]), Bt, 32, DT, PD, DT, PD);
#endif

  v8f acc[16] = {};
  v16bf a_cur = load_tile_a(A, PD, lane);
  const int NSTEP = PD / 32;                   // 64
  for (int i = 0; i < NSTEP; ++i) {
    const int cur = i & 1;
#if GPA_TDM
    if (wv_s == 0) {
      if (i + 1 < NSTEP) {
        tdm_load_2d(lds_off(&wbuf[cur ^ 1][0]), Bt + (i + 1) * 32, 32, DT, PD, DT, PD);
        TDM_WAIT(1);                           // slab i complete, slab i+1 in flight
      } else {
        TDM_WAIT(0);                           // final slab complete
      }
    }
#else
    coop_copy_rows(&wbuf[cur][0], Bt + i * 32, DT, 32, PD, tid, 256);
#endif
    __syncthreads();                           // slab i visible to all waves

    v16bf a = a_cur;
    if (i + 1 < NSTEP) a_cur = load_tile_a(A + (i + 1) * 32, PD, lane);

    const __bf16* bs = &wbuf[cur][0];          // row d: 32 contiguous K elements
#pragma unroll
    for (int t = 0; t < 16; ++t) {
      v16bf bm = load_tile_b(bs + (t * 16) * 32, 32, lane);
      acc[t] = wmma_bf16(a, bm, acc[t]);
    }
    __syncthreads();                           // all readers done before buffer reuse
  }

  const float* bias = (o == 0) ? bqv : (o == 1) ? bkv : bvv;
  const int pcol = lane & 15;
  const int prow = (lane >> 4) << 3;
  if (o < 2) {
    __bf16* dst = (o == 0) ? qb : kb;
#pragma unroll
    for (int t = 0; t < 16; ++t) {
      const int d = t * 16 + pcol;
      const float bv = bias[d];
#pragma unroll
      for (int j = 0; j < 8; ++j)
        dst[(size_t)(m0 + prow + j) * DT + d] = (__bf16)(acc[t][j] + bv);
    }
  } else {
    const int bidx = m0 >> 11;
    const int mloc = m0 & (NP - 1);
#pragma unroll
    for (int t = 0; t < 16; ++t) {
      const int d = t * 16 + pcol;
      const float bv = bias[d];
      v8bf pk;
#pragma unroll
      for (int j = 0; j < 8; ++j) pk[j] = (__bf16)(acc[t][j] + bv);
      *reinterpret_cast<v8bf*>(vT + ((size_t)bidx * DT + d) * NP + mloc + prow) = pk;
    }
  }
}

// ---------------------------------------------------------------- single-pass gated attention
// Block: 8 waves = 8 query tiles of the SAME batch; k/v slabs (16 KB each) shared via
// double-buffered TDM loads into LDS; softmax needs no max-tracking (gate in (0,1/16)).
__global__ __launch_bounds__(256) void gpa_attn_kernel(
    const __bf16* __restrict__ qb,   // [MTOT, DT]
    const __bf16* __restrict__ kb,   // [MTOT, DT]
    const __bf16* __restrict__ vT,   // [BATCH][DT][NP]
    float* __restrict__ out)         // [BATCH, NP, DT] f32
{
  __shared__ __bf16 kbuf[2][32 * DT];          // keys: 32 rows x 256 d
  __shared__ __bf16 vbuf[2][DT * 32];          // vT:  256 d-rows x 32 m
  __shared__ __bf16 ptile[8][16 * 32];         // per-wave P transpose buffer
  const int tid  = threadIdx.x;
  const int lane = tid & 31;
  const int wv   = tid >> 5;
  const int wv_s = __builtin_amdgcn_readfirstlane(tid) >> 5;
  const int tile = blockIdx.x * 8 + wv;        // 0..1023
  const int b    = tile >> 7;                  // uniform per block (8 | 128)
  const int m0   = (tile & 127) << 4;

  const __bf16* qrow = qb + (size_t)(b * NP + m0) * DT;
  const __bf16* kbat = kb + (size_t)b * NP * DT;
  const __bf16* vbat = vT + (size_t)b * DT * NP;

  v16bf qa[8];
#pragma unroll
  for (int t = 0; t < 8; ++t) qa[t] = load_tile_a(qrow + t * 32, DT, lane);

  v16bf onesb;
#pragma unroll
  for (int i = 0; i < 16; ++i) onesb[i] = (__bf16)1.0f;

  v8f acc[16] = {};
  v8f rsum = {};

  __bf16* myp = ptile[wv];
  const int prow = (lane >> 4) << 3;
  const int pcol = lane & 15;

#if GPA_TDM
  if (wv_s == 0) {
    tdm_load_2d(lds_off(&kbuf[0][0]), kbat, 32 * DT, 1, NP * DT, 1, NP * DT); // contiguous slab
    tdm_load_2d(lds_off(&vbuf[0][0]), vbat, 32, DT, NP, DT, NP);
  }
#endif

  const int NSTEP = NP / 32;                   // 64
  for (int i = 0; i < NSTEP; ++i) {
    const int cur = i & 1;
#if GPA_TDM
    if (wv_s == 0) {
      if (i + 1 < NSTEP) {
        tdm_load_2d(lds_off(&kbuf[cur ^ 1][0]), kbat + (size_t)(i + 1) * 32 * DT,
                    32 * DT, 1, NP * DT, 1, NP * DT);
        tdm_load_2d(lds_off(&vbuf[cur ^ 1][0]), vbat + (i + 1) * 32, 32, DT, NP, DT, NP);
        TDM_WAIT(2);                           // slab-pair i complete, pair i+1 in flight
      } else {
        TDM_WAIT(0);                           // final pair complete
      }
    }
#else
    coop_copy_rows(&kbuf[cur][0], kbat + (size_t)i * 32 * DT, 32, DT, DT, tid, 256);
    coop_copy_rows(&vbuf[cur][0], vbat + i * 32, DT, 32, NP, tid, 256);
#endif
    __syncthreads();

    const __bf16* ks = &kbuf[cur][0];
    const __bf16* vs = &vbuf[cur][0];

    // S[16x32] = q_tile @ k_slab^T   (k rows contiguous in d -> N-major B)
    v8f s0 = {}, s1 = {};
#pragma unroll
    for (int t = 0; t < 8; ++t) {
      v16bf b0 = load_tile_b(ks + t * 32, DT, lane);
      v16bf b1 = load_tile_b(ks + 16 * DT + t * 32, DT, lane);
      s0 = wmma_bf16(qa[t], b0, s0);
      s1 = wmma_bf16(qa[t], b1, s1);
    }

    // P = exp(sigmoid(S)/16): C-layout -> LDS 16x32 row-major -> reload in A-layout
#pragma unroll
    for (int j = 0; j < 8; ++j) {
      myp[(j + prow) * 32 + pcol]      = (__bf16)gate(s0[j]);
      myp[(j + prow) * 32 + 16 + pcol] = (__bf16)gate(s1[j]);
    }
    v16bf pa = load_tile_a(myp, 32, lane);     // same-wave DS ordering

    rsum = wmma_bf16(pa, onesb, rsum);         // rowsums via P @ ones

#pragma unroll
    for (int t = 0; t < 16; ++t) {
      v16bf vb = load_tile_b(vs + (t * 16) * 32, 32, lane);
      acc[t] = wmma_bf16(pa, vb, acc[t]);
    }
    __syncthreads();                           // readers done before buffer reuse
  }

  float inv[8];
#pragma unroll
  for (int j = 0; j < 8; ++j) inv[j] = __frcp_rn(rsum[j]);

  float* obase = out + (size_t)(b * NP + m0) * DT;
#pragma unroll
  for (int t = 0; t < 16; ++t) {
#pragma unroll
    for (int j = 0; j < 8; ++j)
      obase[(size_t)(prow + j) * DT + t * 16 + pcol] = acc[t][j] * inv[j];
  }
}

// ---------------------------------------------------------------- host launch
extern "C" void kernel_launch(void* const* d_in, const int* in_sizes, int n_in,
                              void* d_out, int out_size, void* d_ws, size_t ws_size,
                              hipStream_t stream) {
  const float* x  = (const float*)d_in[0];
  const float* Wq = (const float*)d_in[1];
  const float* bq = (const float*)d_in[2];
  const float* Wk = (const float*)d_in[3];
  const float* bk = (const float*)d_in[4];
  const float* Wv = (const float*)d_in[5];
  const float* bv = (const float*)d_in[6];
  float* out = (float*)d_out;

  char* ws = (char*)d_ws;
  __bf16* xb = (__bf16*)(ws);                  // 67108864 B
  __bf16* wt = (__bf16*)(ws + 67108864);       //  3145728 B
  __bf16* qb = (__bf16*)(ws + 70254592);       //  8388608 B
  __bf16* kb = (__bf16*)(ws + 78643200);       //  8388608 B
  __bf16* vT = (__bf16*)(ws + 87031808);       //  8388608 B (total ~95.4 MB)

  const int n4 = (MTOT * PD) / 4;
  gpa_cvt_kernel<<<n4 / 256, 256, 0, stream>>>((const float4*)x, (v4bf*)xb, n4);

  gpa_wtrans_kernel<<<(DT * PD) / 256, 256, 0, stream>>>(Wq, wt);
  gpa_wtrans_kernel<<<(DT * PD) / 256, 256, 0, stream>>>(Wk, wt + (size_t)DT * PD);
  gpa_wtrans_kernel<<<(DT * PD) / 256, 256, 0, stream>>>(Wv, wt + (size_t)2 * DT * PD);

  gpa_qkv_kernel<<<384, 256, 0, stream>>>(xb, wt, bq, bk, bv, qb, kb, vT);  // 3072 waves
  gpa_attn_kernel<<<128, 256, 0, stream>>>(qb, kb, vT, out);                // 1024 waves
}
